// IterRefEmbedding_29420525977870
// MI455X (gfx1250) — compile-verified, matrix operand order checked by reading, not run on hardware
//
#include <hip/hip_runtime.h>

#define DDIM   1024
#define BATCH  1024
#define NSUP   24
#define NSTEPS 3
#define EPSV   1e-8f

typedef __attribute__((ext_vector_type(16))) _Float16 v16h;
typedef __attribute__((ext_vector_type(4)))  _Float16 v4h;
typedef __attribute__((ext_vector_type(8)))  float    v8f;

__device__ __forceinline__ float sigmoidf_(float x) {
    return 1.0f / (1.0f + __expf(-x));
}

// ---------------------------------------------------------------------------
// Fused LSTM-gates GEMM (WMMA f16 -> f32) + LSTM pointwise epilogue.
// gates[m, g*D + n] = sum_k A[m,k] * W[g*D+n, k] + bih + bhh
//   A = [h_old (D) | lin (D)] against Wih (K = 2D), then h_old against Whh (K = D)
// A operand is pre-staged in f16 by the producing kernels, so the inner loop
// is pure b128 loads + WMMA (no cvt).  The K loop is split into three
// statically-bounded phases so the A source pointer is loop-invariant
// (no per-iteration cndmask address selects).
// Block: 128 threads = 4 waves. Block tile: 32 rows x 64 cols.
// Wave w owns cols [n0 + 16w, +16) for ALL 4 gate groups and 2 M-subtiles:
// 8 WMMAs per 32-K step sharing 4 B fragments -> high MMA density, and the
// LSTM nonlinearity fuses entirely in registers (gates never materialized).
// ---------------------------------------------------------------------------
__global__ __launch_bounds__(128)
void lstm_gemm_wmma(const _Float16* __restrict__ h16_old, // [M, D] f16
                    const _Float16* __restrict__ lin16,   // [M, D] f16
                    const _Float16* __restrict__ Wih,     // [4D, 2D] f16
                    const _Float16* __restrict__ Whh,     // [4D, D]  f16
                    const float* __restrict__ bih,
                    const float* __restrict__ bhh,
                    const float* __restrict__ c_old,
                    float* __restrict__ h_new,
                    float* __restrict__ c_new,
                    _Float16* __restrict__ h16_new)
{
    const int wv   = threadIdx.x >> 5;   // wave 0..3
    const int lane = threadIdx.x & 31;
    const int lrow = lane & 15;
    const int kh   = lane >> 4;          // K-half select (0: K 0..15, 1: K 16..31)
    const int kofs = kh << 4;

    const int m0 = blockIdx.x << 5;                       // row tile base (32 rows)
    const int n  = (blockIdx.y << 6) + (wv << 4) + lrow;  // col within D (this lane)

    const _Float16* __restrict__ Ah0 = h16_old + (size_t)(m0 + lrow) * DDIM + kofs;
    const _Float16* __restrict__ Al0 = lin16   + (size_t)(m0 + lrow) * DDIM + kofs;
    const _Float16* __restrict__ Ah1 = h16_old + (size_t)(m0 + 16 + lrow) * DDIM + kofs;
    const _Float16* __restrict__ Al1 = lin16   + (size_t)(m0 + 16 + lrow) * DDIM + kofs;

    const _Float16* __restrict__ Wih_r = Wih + (size_t)n * (2 * DDIM) + kofs;
    const _Float16* __restrict__ Whh_r = Whh + (size_t)n * DDIM + kofs;

    v8f ai0 = {}, af0 = {}, ag0 = {}, ao0 = {};
    v8f ai1 = {}, af1 = {}, ag1 = {}, ao1 = {};

    // ---- Phase 1a: K in [0, D): A = h_old, B = Wih cols [0, D) -------------
    for (int k0 = 0; k0 < DDIM; k0 += 32) {
        const v16h a0 = *(const v16h*)(Ah0 + k0);
        const v16h a1 = *(const v16h*)(Ah1 + k0);
        const _Float16* wp = Wih_r + k0;
        const v16h bi = *(const v16h*)(wp);
        const v16h bf = *(const v16h*)(wp + (size_t)DDIM * (2 * DDIM));
        const v16h bg = *(const v16h*)(wp + (size_t)2 * DDIM * (2 * DDIM));
        const v16h bo = *(const v16h*)(wp + (size_t)3 * DDIM * (2 * DDIM));
        __builtin_prefetch((const void*)(wp + 32), 0, 0);
        ai0 = __builtin_amdgcn_wmma_f32_16x16x32_f16(false, a0, false, bi, (short)0, ai0, false, false);
        ai1 = __builtin_amdgcn_wmma_f32_16x16x32_f16(false, a1, false, bi, (short)0, ai1, false, false);
        af0 = __builtin_amdgcn_wmma_f32_16x16x32_f16(false, a0, false, bf, (short)0, af0, false, false);
        af1 = __builtin_amdgcn_wmma_f32_16x16x32_f16(false, a1, false, bf, (short)0, af1, false, false);
        ag0 = __builtin_amdgcn_wmma_f32_16x16x32_f16(false, a0, false, bg, (short)0, ag0, false, false);
        ag1 = __builtin_amdgcn_wmma_f32_16x16x32_f16(false, a1, false, bg, (short)0, ag1, false, false);
        ao0 = __builtin_amdgcn_wmma_f32_16x16x32_f16(false, a0, false, bo, (short)0, ao0, false, false);
        ao1 = __builtin_amdgcn_wmma_f32_16x16x32_f16(false, a1, false, bo, (short)0, ao1, false, false);
    }

    // ---- Phase 1b: K in [D, 2D): A = lin, B = Wih cols [D, 2D) -------------
    for (int k0 = 0; k0 < DDIM; k0 += 32) {
        const v16h a0 = *(const v16h*)(Al0 + k0);
        const v16h a1 = *(const v16h*)(Al1 + k0);
        const _Float16* wp = Wih_r + DDIM + k0;
        const v16h bi = *(const v16h*)(wp);
        const v16h bf = *(const v16h*)(wp + (size_t)DDIM * (2 * DDIM));
        const v16h bg = *(const v16h*)(wp + (size_t)2 * DDIM * (2 * DDIM));
        const v16h bo = *(const v16h*)(wp + (size_t)3 * DDIM * (2 * DDIM));
        __builtin_prefetch((const void*)(wp + 32), 0, 0);
        ai0 = __builtin_amdgcn_wmma_f32_16x16x32_f16(false, a0, false, bi, (short)0, ai0, false, false);
        ai1 = __builtin_amdgcn_wmma_f32_16x16x32_f16(false, a1, false, bi, (short)0, ai1, false, false);
        af0 = __builtin_amdgcn_wmma_f32_16x16x32_f16(false, a0, false, bf, (short)0, af0, false, false);
        af1 = __builtin_amdgcn_wmma_f32_16x16x32_f16(false, a1, false, bf, (short)0, af1, false, false);
        ag0 = __builtin_amdgcn_wmma_f32_16x16x32_f16(false, a0, false, bg, (short)0, ag0, false, false);
        ag1 = __builtin_amdgcn_wmma_f32_16x16x32_f16(false, a1, false, bg, (short)0, ag1, false, false);
        ao0 = __builtin_amdgcn_wmma_f32_16x16x32_f16(false, a0, false, bo, (short)0, ao0, false, false);
        ao1 = __builtin_amdgcn_wmma_f32_16x16x32_f16(false, a1, false, bo, (short)0, ao1, false, false);
    }

    // ---- Phase 2: K = D over Whh, A = h_old --------------------------------
    for (int k0 = 0; k0 < DDIM; k0 += 32) {
        const v16h a0 = *(const v16h*)(Ah0 + k0);
        const v16h a1 = *(const v16h*)(Ah1 + k0);
        const _Float16* wp = Whh_r + k0;
        const v16h bi = *(const v16h*)(wp);
        const v16h bf = *(const v16h*)(wp + (size_t)DDIM * DDIM);
        const v16h bg = *(const v16h*)(wp + (size_t)2 * DDIM * DDIM);
        const v16h bo = *(const v16h*)(wp + (size_t)3 * DDIM * DDIM);
        __builtin_prefetch((const void*)(wp + 32), 0, 0);
        ai0 = __builtin_amdgcn_wmma_f32_16x16x32_f16(false, a0, false, bi, (short)0, ai0, false, false);
        ai1 = __builtin_amdgcn_wmma_f32_16x16x32_f16(false, a1, false, bi, (short)0, ai1, false, false);
        af0 = __builtin_amdgcn_wmma_f32_16x16x32_f16(false, a0, false, bf, (short)0, af0, false, false);
        af1 = __builtin_amdgcn_wmma_f32_16x16x32_f16(false, a1, false, bf, (short)0, af1, false, false);
        ag0 = __builtin_amdgcn_wmma_f32_16x16x32_f16(false, a0, false, bg, (short)0, ag0, false, false);
        ag1 = __builtin_amdgcn_wmma_f32_16x16x32_f16(false, a1, false, bg, (short)0, ag1, false, false);
        ao0 = __builtin_amdgcn_wmma_f32_16x16x32_f16(false, a0, false, bo, (short)0, ao0, false, false);
        ao1 = __builtin_amdgcn_wmma_f32_16x16x32_f16(false, a1, false, bo, (short)0, ao1, false, false);
    }

    // ---- LSTM pointwise epilogue (fused in registers) ----------------------
    const float b_i = bih[n]            + bhh[n];
    const float b_f = bih[DDIM + n]     + bhh[DDIM + n];
    const float b_g = bih[2 * DDIM + n] + bhh[2 * DDIM + n];
    const float b_o = bih[3 * DDIM + n] + bhh[3 * DDIM + n];

    #pragma unroll
    for (int r = 0; r < 8; ++r) {
        // subtile 0: C/D layout M = r + 8*(lane>=16)
        {
            const int mr = m0 + r + (kh << 3);
            const size_t idx = (size_t)mr * DDIM + n;
            const float iv = sigmoidf_(ai0[r] + b_i);
            const float fv = sigmoidf_(af0[r] + b_f);
            const float gv = tanhf(ag0[r] + b_g);
            const float ov = sigmoidf_(ao0[r] + b_o);
            const float cn = fv * c_old[idx] + iv * gv;
            const float hv = ov * tanhf(cn);
            c_new[idx] = cn;
            h_new[idx] = hv;
            h16_new[idx] = (_Float16)hv;
        }
        // subtile 1
        {
            const int mr = m0 + 16 + r + (kh << 3);
            const size_t idx = (size_t)mr * DDIM + n;
            const float iv = sigmoidf_(ai1[r] + b_i);
            const float fv = sigmoidf_(af1[r] + b_f);
            const float gv = tanhf(ag1[r] + b_g);
            const float ov = sigmoidf_(ao1[r] + b_o);
            const float cn = fv * c_old[idx] + iv * gv;
            const float hv = ov * tanhf(cn);
            c_new[idx] = cn;
            h_new[idx] = hv;
            h16_new[idx] = (_Float16)hv;
        }
    }
}

// ---------------------------------------------------------------------------
// Support attention: one block per (b, n).  x = z[b,n] + s_refine[b,n];
// sims over 24 support rows (cosine), softmax, lin_s[b,n] = att @ support[b].
// Also emits the f16 copy of lin_s consumed by the WMMA GEMM.
// ---------------------------------------------------------------------------
__global__ __launch_bounds__(256)
void attn_s_kernel(const float* __restrict__ support,
                   const float* __restrict__ z,
                   const float* __restrict__ hs_old,
                   const float* __restrict__ inv_ns,
                   float* __restrict__ z_next,
                   _Float16* __restrict__ z_next16)
{
    const int b = blockIdx.x;
    const int nq = blockIdx.y;
    const int t = threadIdx.x;
    const int lane = t & 31, wrp = t >> 5;
    const int d0 = t << 2;

    const size_t rowq = ((size_t)b * NSUP + nq) * DDIM;
    const float4 zr = *(const float4*)(z + rowq + d0);
    const float4 hr = *(const float4*)(hs_old + rowq + d0);
    const float x0 = zr.x + hr.x, x1 = zr.y + hr.y, x2 = zr.z + hr.z, x3 = zr.w + hr.w;

    const float* sb = support + (size_t)b * NSUP * DDIM;

    float p[NSUP + 1];
    p[NSUP] = x0 * x0 + x1 * x1 + x2 * x2 + x3 * x3;
    #pragma unroll
    for (int m = 0; m < NSUP; ++m) {
        float4 s = *(const float4*)(sb + (size_t)m * DDIM + d0);
        p[m] = x0 * s.x + x1 * s.y + x2 * s.z + x3 * s.w;
    }

    __shared__ float red[8][NSUP + 1];
    #pragma unroll
    for (int j = 0; j <= NSUP; ++j) {
        float v = p[j];
        #pragma unroll
        for (int o = 16; o > 0; o >>= 1) v += __shfl_xor(v, o, 32);
        if (lane == 0) red[wrp][j] = v;
    }
    __syncthreads();

    float invx;
    {
        float s = 0.f;
        #pragma unroll
        for (int w = 0; w < 8; ++w) s += red[w][NSUP];
        invx = 1.0f / (sqrtf(s) + EPSV);
    }
    float att[NSUP];
    float mx = -1e30f;
    #pragma unroll
    for (int m = 0; m < NSUP; ++m) {
        float dm = 0.f;
        #pragma unroll
        for (int w = 0; w < 8; ++w) dm += red[w][m];
        const float sim = dm * invx * inv_ns[b * NSUP + m];
        att[m] = sim;
        mx = fmaxf(mx, sim);
    }
    float se = 0.f;
    #pragma unroll
    for (int m = 0; m < NSUP; ++m) { att[m] = __expf(att[m] - mx); se += att[m]; }
    const float inv_se = 1.0f / se;

    float o0 = 0.f, o1 = 0.f, o2 = 0.f, o3 = 0.f;
    #pragma unroll
    for (int m = 0; m < NSUP; ++m) {
        float4 s = *(const float4*)(sb + (size_t)m * DDIM + d0);
        const float a = att[m] * inv_se;
        o0 += a * s.x; o1 += a * s.y; o2 += a * s.z; o3 += a * s.w;
    }
    float4 out; out.x = o0; out.y = o1; out.z = o2; out.w = o3;
    *(float4*)(z_next + rowq + d0) = out;
    v4h out16; out16[0] = (_Float16)o0; out16[1] = (_Float16)o1;
    out16[2] = (_Float16)o2; out16[3] = (_Float16)o3;
    *(v4h*)(z_next16 + rowq + d0) = out16;
}

// ---------------------------------------------------------------------------
// Query attention: one block per b.  x = query[b] + q_refine[b];
// keys/values are the 24 z rows (norms computed in-kernel).
// ---------------------------------------------------------------------------
__global__ __launch_bounds__(256)
void attn_q_kernel(const float* __restrict__ query,
                   const float* __restrict__ hq_old,
                   const float* __restrict__ z,
                   float* __restrict__ lin_q,
                   _Float16* __restrict__ lin_q16)
{
    const int b = blockIdx.x;
    const int t = threadIdx.x;
    const int lane = t & 31, wrp = t >> 5;
    const int d0 = t << 2;

    const float4 q4 = *(const float4*)(query + (size_t)b * DDIM + d0);
    const float4 h4 = *(const float4*)(hq_old + (size_t)b * DDIM + d0);
    const float x0 = q4.x + h4.x, x1 = q4.y + h4.y, x2 = q4.z + h4.z, x3 = q4.w + h4.w;

    const float* zb = z + (size_t)b * NSUP * DDIM;

    float p[2 * NSUP + 1];
    p[2 * NSUP] = x0 * x0 + x1 * x1 + x2 * x2 + x3 * x3;
    #pragma unroll
    for (int m = 0; m < NSUP; ++m) {
        float4 s = *(const float4*)(zb + (size_t)m * DDIM + d0);
        p[m]        = x0 * s.x + x1 * s.y + x2 * s.z + x3 * s.w;
        p[NSUP + m] = s.x * s.x + s.y * s.y + s.z * s.z + s.w * s.w;
    }

    __shared__ float red[8][2 * NSUP + 1];
    #pragma unroll
    for (int j = 0; j <= 2 * NSUP; ++j) {
        float v = p[j];
        #pragma unroll
        for (int o = 16; o > 0; o >>= 1) v += __shfl_xor(v, o, 32);
        if (lane == 0) red[wrp][j] = v;
    }
    __syncthreads();

    float invx;
    {
        float s = 0.f;
        #pragma unroll
        for (int w = 0; w < 8; ++w) s += red[w][2 * NSUP];
        invx = 1.0f / (sqrtf(s) + EPSV);
    }
    float att[NSUP];
    float mx = -1e30f;
    #pragma unroll
    for (int m = 0; m < NSUP; ++m) {
        float dm = 0.f, zn = 0.f;
        #pragma unroll
        for (int w = 0; w < 8; ++w) { dm += red[w][m]; zn += red[w][NSUP + m]; }
        const float sim = dm * invx / (sqrtf(zn) + EPSV);
        att[m] = sim;
        mx = fmaxf(mx, sim);
    }
    float se = 0.f;
    #pragma unroll
    for (int m = 0; m < NSUP; ++m) { att[m] = __expf(att[m] - mx); se += att[m]; }
    const float inv_se = 1.0f / se;

    float o0 = 0.f, o1 = 0.f, o2 = 0.f, o3 = 0.f;
    #pragma unroll
    for (int m = 0; m < NSUP; ++m) {
        float4 s = *(const float4*)(zb + (size_t)m * DDIM + d0);
        const float a = att[m] * inv_se;
        o0 += a * s.x; o1 += a * s.y; o2 += a * s.z; o3 += a * s.w;
    }
    float4 out; out.x = o0; out.y = o1; out.z = o2; out.w = o3;
    *(float4*)(lin_q + (size_t)b * DDIM + d0) = out;
    v4h out16; out16[0] = (_Float16)o0; out16[1] = (_Float16)o1;
    out16[2] = (_Float16)o2; out16[3] = (_Float16)o3;
    *(v4h*)(lin_q16 + (size_t)b * DDIM + d0) = out16;
}

// ------------------------------ prep kernels -------------------------------
__global__ void cvt_f16_kernel(const float* __restrict__ src, _Float16* __restrict__ dst, size_t n) {
    size_t i = (size_t)blockIdx.x * blockDim.x + threadIdx.x;
    const size_t st = (size_t)gridDim.x * blockDim.x;
    for (; i < n; i += st) dst[i] = (_Float16)src[i];
}

__global__ void build_support_kernel(const float* __restrict__ act,
                                     const float* __restrict__ inact,
                                     float* __restrict__ support) {
    const size_t n = (size_t)BATCH * NSUP * DDIM;
    size_t i = (size_t)blockIdx.x * blockDim.x + threadIdx.x;
    const size_t st = (size_t)gridDim.x * blockDim.x;
    for (; i < n; i += st) {
        const int d = (int)(i % DDIM);
        const size_t r = i / DDIM;
        const int m = (int)(r % NSUP);
        const size_t b = r / NSUP;
        support[i] = (m < 12) ? act[(b * 12 + m) * DDIM + d]
                              : inact[(b * 12 + (m - 12)) * DDIM + d];
    }
}

__global__ __launch_bounds__(256)
void rownorm_kernel(const float* __restrict__ rows, float* __restrict__ inv_out) {
    const int r = blockIdx.x;
    const int t = threadIdx.x;
    const int lane = t & 31, wrp = t >> 5;
    const float* p = rows + (size_t)r * DDIM;
    float s = 0.f;
    for (int d = t; d < DDIM; d += 256) { const float v = p[d]; s += v * v; }
    #pragma unroll
    for (int o = 16; o > 0; o >>= 1) s += __shfl_xor(s, o, 32);
    __shared__ float red[8];
    if (lane == 0) red[wrp] = s;
    __syncthreads();
    if (t == 0) {
        float tot = 0.f;
        #pragma unroll
        for (int w = 0; w < 8; ++w) tot += red[w];
        inv_out[r] = 1.0f / (sqrtf(tot) + EPSV);
    }
}

__global__ void zero_kernel(float* __restrict__ p, size_t n) {
    size_t i = (size_t)blockIdx.x * blockDim.x + threadIdx.x;
    const size_t st = (size_t)gridDim.x * blockDim.x;
    for (; i < n; i += st) p[i] = 0.0f;
}

__global__ void finalize_kernel(const float* __restrict__ query,
                                const float* __restrict__ hq,
                                const float* __restrict__ support,
                                const float* __restrict__ hs,
                                float* __restrict__ out) {
    const size_t QS = (size_t)BATCH * DDIM;
    const size_t AS = (size_t)BATCH * 12 * DDIM;
    const size_t n = QS + 2 * AS;
    size_t i = (size_t)blockIdx.x * blockDim.x + threadIdx.x;
    const size_t st = (size_t)gridDim.x * blockDim.x;
    for (; i < n; i += st) {
        if (i < QS) {
            out[i] = query[i] + hq[i];
        } else if (i < QS + AS) {
            const size_t j = i - QS;
            const size_t b = j / (12 * DDIM);
            const size_t rem = j % (12 * DDIM);
            const size_t m = rem / DDIM, d = rem % DDIM;
            const size_t s = (b * NSUP + m) * DDIM + d;
            out[i] = support[s] + hs[s];
        } else {
            const size_t j = i - QS - AS;
            const size_t b = j / (12 * DDIM);
            const size_t rem = j % (12 * DDIM);
            const size_t m = rem / DDIM + 12, d = rem % DDIM;
            const size_t s = (b * NSUP + m) * DDIM + d;
            out[i] = support[s] + hs[s];
        }
    }
}

// ---------------------------------------------------------------------------
extern "C" void kernel_launch(void* const* d_in, const int* in_sizes, int n_in,
                              void* d_out, int out_size, void* d_ws, size_t ws_size,
                              hipStream_t stream) {
    const float* query = (const float*)d_in[0];
    const float* act   = (const float*)d_in[1];
    const float* inact = (const float*)d_in[2];
    const float* Wih_s = (const float*)d_in[3];
    const float* Whh_s = (const float*)d_in[4];
    const float* bih_s = (const float*)d_in[5];
    const float* bhh_s = (const float*)d_in[6];
    const float* Wih_q = (const float*)d_in[7];
    const float* Whh_q = (const float*)d_in[8];
    const float* bih_q = (const float*)d_in[9];
    const float* bhh_q = (const float*)d_in[10];

    char* ws = (char*)d_ws;
    size_t off = 0;
    auto carve = [&](size_t bytes) -> char* {
        char* p = ws + off;
        off = (off + bytes + 255) & ~(size_t)255;
        return p;
    };

    const size_t SD = (size_t)BATCH * NSUP * DDIM;  // 25.17M elems
    const size_t QD = (size_t)BATCH * DDIM;
    const size_t WIH = (size_t)4 * DDIM * 2 * DDIM; // 8.39M
    const size_t WHH = (size_t)4 * DDIM * DDIM;     // 4.19M

    float* support = (float*)carve(SD * 4);
    float* z_a  = (float*)carve(SD * 4);
    float* z_b  = (float*)carve(SD * 4);
    float* hs_a = (float*)carve(SD * 4);
    float* hs_b = (float*)carve(SD * 4);
    float* cs_a = (float*)carve(SD * 4);
    float* cs_b = (float*)carve(SD * 4);
    float* hq_a = (float*)carve(QD * 4);
    float* hq_b = (float*)carve(QD * 4);
    float* cq_a = (float*)carve(QD * 4);
    float* cq_b = (float*)carve(QD * 4);
    float* lin_q  = (float*)carve(QD * 4);
    float* inv_ns = (float*)carve((size_t)BATCH * NSUP * 4);
    // f16 staging for the WMMA A-operands
    _Float16* hs16_a = (_Float16*)carve(SD * 2);
    _Float16* hs16_b = (_Float16*)carve(SD * 2);
    _Float16* z16    = (_Float16*)carve(SD * 2);
    _Float16* hq16_a = (_Float16*)carve(QD * 2);
    _Float16* hq16_b = (_Float16*)carve(QD * 2);
    _Float16* lq16   = (_Float16*)carve(QD * 2);
    // f16 weights
    _Float16* WihS = (_Float16*)carve(WIH * 2);
    _Float16* WhhS = (_Float16*)carve(WHH * 2);
    _Float16* WihQ = (_Float16*)carve(WIH * 2);
    _Float16* WhhQ = (_Float16*)carve(WHH * 2);

    // ---- one-time prep (re-run every call: deterministic) ------------------
    cvt_f16_kernel<<<2048, 256, 0, stream>>>(Wih_s, WihS, WIH);
    cvt_f16_kernel<<<2048, 256, 0, stream>>>(Whh_s, WhhS, WHH);
    cvt_f16_kernel<<<2048, 256, 0, stream>>>(Wih_q, WihQ, WIH);
    cvt_f16_kernel<<<2048, 256, 0, stream>>>(Whh_q, WhhQ, WHH);
    build_support_kernel<<<4096, 256, 0, stream>>>(act, inact, support);
    rownorm_kernel<<<BATCH * NSUP, 256, 0, stream>>>(support, inv_ns);
    zero_kernel<<<4096, 256, 0, stream>>>(hs_a, SD);
    zero_kernel<<<4096, 256, 0, stream>>>(cs_a, SD);
    zero_kernel<<<256, 256, 0, stream>>>(hq_a, QD);
    zero_kernel<<<256, 256, 0, stream>>>(cq_a, QD);
    zero_kernel<<<2048, 256, 0, stream>>>((float*)hs16_a, SD / 2);  // 0x0000 == +0.0h
    zero_kernel<<<128, 256, 0, stream>>>((float*)hq16_a, QD / 2);

    const float* z_cur = support;   // step 0: z == support
    float* z_nxt = z_a;
    float    *hs_o = hs_a, *hs_n = hs_b, *cs_o = cs_a, *cs_n = cs_b;
    float    *hq_o = hq_a, *hq_n = hq_b, *cq_o = cq_a, *cq_n = cq_b;
    _Float16 *hs16_o = hs16_a, *hs16_n = hs16_b;
    _Float16 *hq16_o = hq16_a, *hq16_n = hq16_b;

    for (int step = 0; step < NSTEPS; ++step) {
        attn_q_kernel<<<BATCH, 256, 0, stream>>>(query, hq_o, z_cur, lin_q, lq16);
        attn_s_kernel<<<dim3(BATCH, NSUP), 256, 0, stream>>>(support, z_cur, hs_o, inv_ns, z_nxt, z16);
        // support LSTM: M = B*24, lin_s = z_nxt (f16 copy in z16)
        lstm_gemm_wmma<<<dim3((BATCH * NSUP) / 32, DDIM / 64), 128, 0, stream>>>(
            hs16_o, z16, WihS, WhhS, bih_s, bhh_s, cs_o, hs_n, cs_n, hs16_n);
        // query LSTM: M = B
        lstm_gemm_wmma<<<dim3(BATCH / 32, DDIM / 64), 128, 0, stream>>>(
            hq16_o, lq16, WihQ, WhhQ, bih_q, bhh_q, cq_o, hq_n, cq_n, hq16_n);
        { float* t = hs_o; hs_o = hs_n; hs_n = t; }
        { float* t = cs_o; cs_o = cs_n; cs_n = t; }
        { float* t = hq_o; hq_o = hq_n; hq_n = t; }
        { float* t = cq_o; cq_o = cq_n; cq_n = t; }
        { _Float16* t = hs16_o; hs16_o = hs16_n; hs16_n = t; }
        { _Float16* t = hq16_o; hq16_o = hq16_n; hq16_n = t; }
        z_cur = z_nxt;
        z_nxt = (z_nxt == z_a) ? z_b : z_a;
    }

    finalize_kernel<<<4096, 256, 0, stream>>>(query, hq_o, support, hs_o, (float*)d_out);

    (void)in_sizes; (void)n_in; (void)out_size; (void)ws_size;
}
